// TransformerBlock_62062277427683
// MI455X (gfx1250) — matrix operand order repaired and checked by reading, not compiled, not run on hardware
//
#include <hip/hip_runtime.h>
#include <hip/hip_bf16.h>

// ---------------------------------------------------------------------------
// Transformer block forward for gfx1250 (MI455X), wave32 + WMMA bf16.
// B=4, S=2048, D=768, H=12, DK=64, DFF=3072.  ~167 GFLOP, compute-bound.
// All GEMM-shaped work on v_wmma_f32_16x16x32_bf16 (fp32 accum); LayerNorm /
// softmax / GELU / residuals in fp32.  Attention fused flash-style with a
// shift-free exact softmax (row sums via a WMMA against a ones matrix — no
// cross-lane shuffles in the hot loop).
// ---------------------------------------------------------------------------

typedef unsigned int  u32;
typedef unsigned short u16;
typedef __attribute__((ext_vector_type(16))) __bf16 v16bf;
typedef __attribute__((ext_vector_type(8)))  float  v8f;

#define CB 4
#define CS 2048
#define CD 768
#define CH 12
#define CDK 64
#define CDFF 3072
#define NTOK (CB*CS)        // 8192

union Frag16 { v16bf v; u32 u[8]; };

static __device__ __forceinline__ u16 f2bf(float f) {
  union { __hip_bfloat16 h; u16 u; } c;
  c.h = __float2bfloat16(f);
  return c.u;
}

static __device__ __forceinline__ v8f wmma_bf16(const Frag16& a, const Frag16& b, v8f c) {
  // D = A(16x32 bf16) * B(32x16 bf16) + C(16x16 f32)
  return __builtin_amdgcn_wmma_f32_16x16x32_bf16(false, a.v, false, b.v,
                                                 (short)0, c, false, false);
}

// A-matrix (16x32, 16-bit) per-lane K index for VGPR i (ISA 7.12.2)
static __device__ __forceinline__ int a_kk(int i, int half) {
  return (i < 4) ? (2*i + 8*half) : (16 + 2*(i-4) + 8*half);
}
// B-matrix (32x16, 16-bit): same K order as A-matrix
static __device__ __forceinline__ int b_kk(int i, int half) {
  return (i < 4) ? (2*i + 8*half) : (16 + 2*(i-4) + 8*half);
}

// ---------------------------------------------------------------------------
// Weight transpose + f32->bf16:  W[K][N] f32  ->  WT[N][K] bf16
// ---------------------------------------------------------------------------
__global__ __launch_bounds__(256) void wtr_kernel(const float* __restrict__ W,
                                                  u16* __restrict__ WT,
                                                  int K, int N) {
  __shared__ float t[32][33];
  const int bn = blockIdx.x * 32;   // N
  const int bk = blockIdx.y * 32;   // K
  const int tx = threadIdx.x & 31;
  const int ty = threadIdx.x >> 5;  // 0..7
  #pragma unroll
  for (int r = 0; r < 32; r += 8)
    t[ty + r][tx] = W[(size_t)(bk + ty + r) * N + bn + tx];
  __syncthreads();
  #pragma unroll
  for (int r = 0; r < 32; r += 8)
    WT[(size_t)(bn + ty + r) * K + bk + tx] = f2bf(t[tx][ty + r]);
}

// ---------------------------------------------------------------------------
// LayerNorm (torch-style: ddof=1 std, eps added to std), f32 in -> bf16 out
// ---------------------------------------------------------------------------
__global__ __launch_bounds__(256) void ln_kernel(const float* __restrict__ x,
                                                 const float* __restrict__ g,
                                                 const float* __restrict__ bta,
                                                 u16* __restrict__ out) {
  const int row = blockIdx.x;                 // token
  const int tid = threadIdx.x;
  const float* xr = x + (size_t)row * CD;
  float vals[3], s = 0.f, ss = 0.f;
  #pragma unroll
  for (int i = 0; i < 3; ++i) {
    float v = xr[tid + 256 * i];
    vals[i] = v; s += v; ss += v * v;
  }
  __shared__ float rs[256], rq[256];
  rs[tid] = s; rq[tid] = ss;
  __syncthreads();
  for (int off = 128; off > 0; off >>= 1) {
    if (tid < off) { rs[tid] += rs[tid + off]; rq[tid] += rq[tid + off]; }
    __syncthreads();
  }
  const float mean = rs[0] * (1.0f / CD);
  const float var  = (rq[0] - (float)CD * mean * mean) * (1.0f / (CD - 1));
  const float rstd = 1.0f / (sqrtf(fmaxf(var, 0.0f)) + 1e-6f);
  u16* orow = out + (size_t)row * CD;
  #pragma unroll
  for (int i = 0; i < 3; ++i) {
    int c = tid + 256 * i;
    orow[c] = f2bf((vals[i] - mean) * rstd * g[c] + bta[c]);
  }
}

// ---------------------------------------------------------------------------
// Tiled bf16 GEMM:  C[M][N] = A[M][K] @ Bt[N][K]^T + bias, fused epilogue.
//   MODE 0: write bf16 scattered to [B,H,S,DK]        (q, k)
//   MODE 1: write bf16 scattered to [B,H,DK,S]        (v transposed)
//   MODE 2: gelu(acc+bias) -> bf16 [M][N]             (FFN1)
//   MODE 3: res[M][N] + acc + bias -> f32 [M][N]      (Wo, FFN2 residual)
// Block = 128 threads (4 waves); tile 128x64, k-step 32; each wave computes
// 32x64 via 8 x v_wmma_f32_16x16x32_bf16 per k-step (B-fragment reuse x2).
// ---------------------------------------------------------------------------
template <int MODE>
__global__ __launch_bounds__(128) void gemm_kernel(const u16* __restrict__ A,
                                                   const u16* __restrict__ Bt,
                                                   const float* __restrict__ bias,
                                                   void* __restrict__ out,
                                                   const float* __restrict__ res,
                                                   int M, int N, int K) {
  __shared__ u16 As[128][32];
  __shared__ u16 Bs[64][32];
  const int tid  = threadIdx.x;
  const int wave = tid >> 5;
  const int lane = tid & 31;
  const int half = lane >> 4;
  const int lm   = lane & 15;
  const int m0   = blockIdx.x * 128;
  const int n0   = blockIdx.y * 64;

  v8f acc[2][4];
  #pragma unroll
  for (int s = 0; s < 2; ++s)
    #pragma unroll
    for (int j = 0; j < 4; ++j) acc[s][j] = {};

  for (int k0 = 0; k0 < K; k0 += 32) {
    // A tile: 128x32 bf16 = 512 x 16B chunks; B tile: 64x32 = 256 chunks.
    #pragma unroll
    for (int r = 0; r < 4; ++r) {
      int c = tid + 128 * r;          // 0..511
      int row = c >> 2, ch = c & 3;
      *reinterpret_cast<uint4*>(&As[row][ch * 8]) =
          *reinterpret_cast<const uint4*>(&A[(size_t)(m0 + row) * K + k0 + ch * 8]);
    }
    #pragma unroll
    for (int r = 0; r < 2; ++r) {
      int c = tid + 128 * r;          // 0..255
      int row = c >> 2, ch = c & 3;
      *reinterpret_cast<uint4*>(&Bs[row][ch * 8]) =
          *reinterpret_cast<const uint4*>(&Bt[(size_t)(n0 + row) * K + k0 + ch * 8]);
    }
    __syncthreads();

    Frag16 af[2];
    #pragma unroll
    for (int s = 0; s < 2; ++s) {
      const int am = 32 * wave + 16 * s + lm;
      #pragma unroll
      for (int i = 0; i < 8; ++i)
        af[s].u[i] = *reinterpret_cast<const u32*>(&As[am][a_kk(i, half)]);
    }
    #pragma unroll
    for (int j = 0; j < 4; ++j) {
      Frag16 bf;
      const int bn = 16 * j + lm;
      #pragma unroll
      for (int i = 0; i < 8; ++i)
        bf.u[i] = *reinterpret_cast<const u32*>(&Bs[bn][b_kk(i, half)]);
      #pragma unroll
      for (int s = 0; s < 2; ++s)
        acc[s][j] = wmma_bf16(af[s], bf, acc[s][j]);
    }
    __syncthreads();
  }

  // Epilogue.  C layout: VGPR i -> row (+16s) m0+32w+16s+i+8*half, col n0+16j+lm.
  #pragma unroll
  for (int s = 0; s < 2; ++s) {
    #pragma unroll
    for (int j = 0; j < 4; ++j) {
      #pragma unroll
      for (int i = 0; i < 8; ++i) {
        const int m = m0 + 32 * wave + 16 * s + i + 8 * half;
        const int n = n0 + 16 * j + lm;
        float v = acc[s][j][i] + bias[n];
        if (MODE == 0 || MODE == 1) {
          const int b = m >> 11, ms = m & (CS - 1);
          const int hh = n >> 6, dk = n & (CDK - 1);
          const size_t off = (MODE == 0)
              ? (((size_t)(b * CH + hh) * CS + ms) * CDK + dk)
              : (((size_t)(b * CH + hh) * CDK + dk) * CS + ms);
          ((u16*)out)[off] = f2bf(v);
        } else if (MODE == 2) {
          const float ge = 0.5f * v * (1.0f + erff(v * 0.70710678118654752f));
          ((u16*)out)[(size_t)m * N + n] = f2bf(ge);
        } else {
          ((float*)out)[(size_t)m * N + n] = res[(size_t)m * N + n] + v;
        }
      }
    }
  }
}

// ---------------------------------------------------------------------------
// Fused flash attention, shift-free exact softmax.
// One wave = one 16-query tile; block = 4 waves, waves fully independent.
// q,k: bf16 [B,H,S,DK];  vT: bf16 [B,H,DK,S];  out: bf16 [NTOK][D].
// Per 32-key block: 4 WMMA scores, mask+scale+exp (exp(min(s,80)); softmax
// shift cancels mathematically, scores are O(1) here), P -> LDS C->A
// relayout, 1 WMMA row-sum against ones, 4 WMMA P@V into fp32 O.
// ---------------------------------------------------------------------------
__global__ __launch_bounds__(128) void fattn_kernel(const u16* __restrict__ q,
                                                    const u16* __restrict__ k,
                                                    const u16* __restrict__ vT,
                                                    const int* __restrict__ mask,
                                                    u16* __restrict__ attnb) {
  __shared__ u16 plds[4][16][32];
  const int tid  = threadIdx.x;
  const int wave = tid >> 5;
  const int lane = tid & 31;
  const int half = lane >> 4;
  const int lm   = lane & 15;

  const int bid = blockIdx.x;       // B*H*(S/64) = 1536
  const int qt  = bid & 31;         // query super-tile (64 q)
  const int bh  = bid >> 5;         // 0..47
  const int b   = bh / CH;
  const int hh  = bh - b * CH;
  const int q0  = qt * 64 + wave * 16;

  const u16* qp = q  + (size_t)bh * CS * CDK;
  const u16* kp = k  + (size_t)bh * CS * CDK;
  const u16* vp = vT + (size_t)bh * CDK * CS;
  const int* mp = mask + b * CS;

  // Preload q A-fragments for the two K-steps over DK=64.
  Frag16 qf[2];
  const int qm = q0 + lm;
  #pragma unroll
  for (int ks = 0; ks < 2; ++ks)
    #pragma unroll
    for (int i = 0; i < 8; ++i)
      qf[ks].u[i] = *reinterpret_cast<const u32*>(
          &qp[(size_t)qm * CDK + a_kk(i, half) + 32 * ks]);

  Frag16 ones;
  #pragma unroll
  for (int i = 0; i < 8; ++i) ones.u[i] = 0x3F803F80u;   // bf16 1.0 pair

  float li[8];
  #pragma unroll
  for (int i = 0; i < 8; ++i) li[i] = 0.0f;
  v8f O[4];
  #pragma unroll
  for (int j = 0; j < 4; ++j) O[j] = {};

  for (int kb = 0; kb < CS / 32; ++kb) {
    const int key0 = kb * 32;
    // scores: two 16x16 C tiles over this 32-key block
    v8f sc[2];
    #pragma unroll
    for (int t = 0; t < 2; ++t) {
      Frag16 kf0, kf1;
      const int key = key0 + 16 * t + lm;
      #pragma unroll
      for (int i = 0; i < 8; ++i) {
        const int kk = b_kk(i, half);
        kf0.u[i] = *reinterpret_cast<const u32*>(&kp[(size_t)key * CDK + kk]);
        kf1.u[i] = *reinterpret_cast<const u32*>(&kp[(size_t)key * CDK + kk + 32]);
      }
      v8f z = {};
      z = wmma_bf16(qf[0], kf0, z);
      z = wmma_bf16(qf[1], kf1, z);
      sc[t] = z;
    }
    // scale by 1/sqrt(64), mask -> exactly -1e9, p = exp(min(s,80))
    // (masked lanes: exp(-1e9) == 0, matching reference semantics)
    #pragma unroll
    for (int t = 0; t < 2; ++t) {
      const bool ok = mp[key0 + 16 * t + lm] != 0;
      #pragma unroll
      for (int i = 0; i < 8; ++i) {
        const float s = ok ? sc[t][i] * 0.125f : -1.0e9f;
        sc[t][i] = __expf(fminf(s, 80.0f));
      }
    }
    // P (16x32) C-layout -> LDS -> A-layout fragment (wave-private tile;
    // LDS is in-order per wave, fence stops compiler reordering)
    #pragma unroll
    for (int t = 0; t < 2; ++t)
      #pragma unroll
      for (int i = 0; i < 8; ++i)
        plds[wave][i + 8 * half][16 * t + lm] = f2bf(sc[t][i]);
    asm volatile("s_wait_dscnt 0" ::: "memory");

    Frag16 pf;
    #pragma unroll
    for (int i = 0; i < 8; ++i)
      pf.u[i] = *reinterpret_cast<const u32*>(&plds[wave][lm][a_kk(i, half)]);

    // row sums of P via WMMA against ones: C[i] = sum_k P[row i+8*half][k]
    {
      v8f z = {};
      z = wmma_bf16(pf, ones, z);
      #pragma unroll
      for (int i = 0; i < 8; ++i) li[i] += z[i];
    }

    // P @ V : B-operand from vT[dk][key] (K pairs contiguous along keys)
    #pragma unroll
    for (int j = 0; j < 4; ++j) {
      Frag16 vf;
      const int dk = 16 * j + lm;
      #pragma unroll
      for (int i = 0; i < 8; ++i)
        vf.u[i] = *reinterpret_cast<const u32*>(
            &vp[(size_t)dk * CS + key0 + b_kk(i, half)]);
      O[j] = wmma_bf16(pf, vf, O[j]);
    }
  }

  // normalize and scatter back to [token][D] bf16 for the Wo GEMM
  float inv[8];
  #pragma unroll
  for (int i = 0; i < 8; ++i) inv[i] = 1.0f / li[i];
  #pragma unroll
  for (int j = 0; j < 4; ++j)
    #pragma unroll
    for (int i = 0; i < 8; ++i) {
      const int srow = q0 + i + 8 * half;
      const int col  = hh * CDK + 16 * j + lm;
      attnb[(size_t)(b * CS + srow) * CD + col] = f2bf(O[j][i] * inv[i]);
    }
}

// ---------------------------------------------------------------------------
// Host launcher
// ---------------------------------------------------------------------------
extern "C" void kernel_launch(void* const* d_in, const int* in_sizes, int n_in,
                              void* d_out, int out_size, void* d_ws, size_t ws_size,
                              hipStream_t stream) {
  (void)in_sizes; (void)n_in; (void)out_size; (void)ws_size;

  const float* x    = (const float*)d_in[0];
  const int*   mask = (const int*)  d_in[1];
  const float* Wq   = (const float*)d_in[2];
  const float* bq   = (const float*)d_in[3];
  const float* Wk   = (const float*)d_in[4];
  const float* bk   = (const float*)d_in[5];
  const float* Wv   = (const float*)d_in[6];
  const float* bv   = (const float*)d_in[7];
  const float* Wo   = (const float*)d_in[8];
  const float* bo   = (const float*)d_in[9];
  const float* W1   = (const float*)d_in[10];
  const float* b1   = (const float*)d_in[11];
  const float* W2   = (const float*)d_in[12];
  const float* b2   = (const float*)d_in[13];
  const float* ln1g = (const float*)d_in[14];
  const float* ln1b = (const float*)d_in[15];
  const float* ln2g = (const float*)d_in[16];
  const float* ln2b = (const float*)d_in[17];
  float* out = (float*)d_out;

  char* p = (char*)d_ws;
  auto alloc = [&](size_t bytes) -> char* {
    char* r = p; p += (bytes + 255) & ~(size_t)255; return r;
  };
  u16* WqT = (u16*)alloc((size_t)CD * CD * 2);
  u16* WkT = (u16*)alloc((size_t)CD * CD * 2);
  u16* WvT = (u16*)alloc((size_t)CD * CD * 2);
  u16* WoT = (u16*)alloc((size_t)CD * CD * 2);
  u16* W1T = (u16*)alloc((size_t)CD * CDFF * 2);
  u16* W2T = (u16*)alloc((size_t)CDFF * CD * 2);
  u16* h1  = (u16*)alloc((size_t)NTOK * CD * 2);
  u16* qb  = (u16*)alloc((size_t)NTOK * CD * 2);
  u16* kb2 = (u16*)alloc((size_t)NTOK * CD * 2);
  u16* vTb = (u16*)alloc((size_t)NTOK * CD * 2);
  u16* atb = (u16*)alloc((size_t)NTOK * CD * 2);
  float* x1 = (float*)alloc((size_t)NTOK * CD * 4);
  u16* h2  = (u16*)alloc((size_t)NTOK * CD * 2);
  u16* a2  = (u16*)alloc((size_t)NTOK * CDFF * 2);

  // weight transposes (f32 [K][N] -> bf16 [N][K])
  wtr_kernel<<<dim3(CD/32,   CD/32),   256, 0, stream>>>(Wq, WqT, CD,   CD);
  wtr_kernel<<<dim3(CD/32,   CD/32),   256, 0, stream>>>(Wk, WkT, CD,   CD);
  wtr_kernel<<<dim3(CD/32,   CD/32),   256, 0, stream>>>(Wv, WvT, CD,   CD);
  wtr_kernel<<<dim3(CD/32,   CD/32),   256, 0, stream>>>(Wo, WoT, CD,   CD);
  wtr_kernel<<<dim3(CDFF/32, CD/32),   256, 0, stream>>>(W1, W1T, CD,   CDFF);
  wtr_kernel<<<dim3(CD/32,   CDFF/32), 256, 0, stream>>>(W2, W2T, CDFF, CD);

  // attention sublayer
  ln_kernel<<<NTOK, 256, 0, stream>>>(x, ln1g, ln1b, h1);
  gemm_kernel<0><<<dim3(NTOK/128, CD/64), 128, 0, stream>>>(h1, WqT, bq, qb,  nullptr, NTOK, CD, CD);
  gemm_kernel<0><<<dim3(NTOK/128, CD/64), 128, 0, stream>>>(h1, WkT, bk, kb2, nullptr, NTOK, CD, CD);
  gemm_kernel<1><<<dim3(NTOK/128, CD/64), 128, 0, stream>>>(h1, WvT, bv, vTb, nullptr, NTOK, CD, CD);
  fattn_kernel<<<CB * CH * (CS / 64), 128, 0, stream>>>(qb, kb2, vTb, mask, atb);
  gemm_kernel<3><<<dim3(NTOK/128, CD/64), 128, 0, stream>>>(atb, WoT, bo, x1, x, NTOK, CD, CD);

  // FFN sublayer
  ln_kernel<<<NTOK, 256, 0, stream>>>(x1, ln2g, ln2b, h2);
  gemm_kernel<2><<<dim3(NTOK/128, CDFF/64), 128, 0, stream>>>(h2, W1T, b1, a2,  nullptr, NTOK, CDFF, CD);
  gemm_kernel<3><<<dim3(NTOK/128, CD/64),   128, 0, stream>>>(a2, W2T, b2, out, x1,      NTOK, CD, CDFF);
}
